// MultiScaleFrameNetwork_9663676416055
// MI455X (gfx1250) — compile-verified, matrix operand order checked by reading, not run on hardware
//
#include <hip/hip_runtime.h>
#include <hip/hip_bf16.h>

typedef __attribute__((ext_vector_type(16))) _Float16 v16h;
typedef __attribute__((ext_vector_type(8)))  _Float16 v8h;
typedef __attribute__((ext_vector_type(4)))  _Float16 v4h;
typedef __attribute__((ext_vector_type(8)))  float    v8f;

#define HID_S  128
#define HID_V  16
#define EDGE_S 64
#define KNN    24
#define KSLOT  32
#define NF     4
#define NPTS   4096
#define BATCH  2
#define EPSF   1e-8f

// ------------------------------------------------------------------
// Pack Ws (Krows x 128, f32 row-major) into f16 WMMA B-fragments.
// Fragment (kc,nt) = 512 halfs; lane L holds 16 halfs at frag*512 + L*16:
//   n    = nt*16 + (L&15)
//   k(j) = kc*32 + ((L>>4)?8:0) + (j&7) + ((j>>3)?16:0)
// Rows k >= Krows are zeroed, so A-tile K-padding never contributes.
// ------------------------------------------------------------------
struct PackArgs { const float* Ws[15]; };

__global__ void pack_ws_kernel(PackArgs pa, _Float16* out)
{
    int mat = blockIdx.x;               // stack*3 + layer, 15 matrices
    int l   = mat % 3;
    int s   = mat / 3;
    int Krows = (l == 0) ? 209 : 144;
    int Kc    = (l == 0) ? 7   : 5;
    int nelem = Kc * 8 * 512;
    const float* W = pa.Ws[mat];
    long base = (long)s * 69632 + (l == 0 ? 0 : (l == 1 ? 28672 : 49152));
    for (int t = threadIdx.x; t < nelem; t += blockDim.x) {
        int frag = t >> 9;
        int r    = t & 511;
        int lane = r >> 4;
        int j    = r & 15;
        int kc   = frag >> 3;
        int nt   = frag & 7;
        int n = nt * 16 + (lane & 15);
        int k = kc * 32 + ((lane >> 4) ? 8 : 0) + (j & 7) + ((j >> 3) ? 16 : 0);
        float v = (k < Krows) ? W[(long)k * 128 + n] : 0.f;
        out[base + t] = (_Float16)v;
    }
}

// ------------------------------------------------------------------
// Centroid of p over N per batch -> pm (B,1,3)
// ------------------------------------------------------------------
__global__ void mean_kernel(const float* p, float* pm)
{
    __shared__ float red[128 * 6];
    int tid = threadIdx.x;
    float a[6] = {0.f, 0.f, 0.f, 0.f, 0.f, 0.f};
    for (int nn = tid; nn < NPTS; nn += 128) {
        for (int b = 0; b < BATCH; ++b) {
            const float* pp = p + ((long)b * NPTS + nn) * 3;
            a[b * 3 + 0] += pp[0];
            a[b * 3 + 1] += pp[1];
            a[b * 3 + 2] += pp[2];
        }
    }
    for (int i = 0; i < 6; ++i) red[tid * 6 + i] = a[i];
    __syncthreads();
    if (tid == 0) {
        for (int i = 0; i < 6; ++i) {
            float s = 0.f;
            for (int t2 = 0; t2 < 128; ++t2) s += red[t2 * 6 + i];
            pm[i] = s / (float)NPTS;
        }
    }
}

// ------------------------------------------------------------------
// kNN (k=24, padded to 32 slots) + RBF edge scalars (f16) + unit dirs.
// One block per query. Matches jax.lax.top_k ordering (d2 asc, index
// tie-break) via 24 argmin-extraction rounds over an LDS d2 table.
// ------------------------------------------------------------------
__global__ void knn_kernel(const float* pc, long c_bstride, int Nc,
                           const float* pq, long q_bstride, int Nq,
                           int* idx_out, _Float16* es_out, float* ev_out)
{
    __shared__ float d2s[4096];
    __shared__ float rd[256];
    __shared__ int   ri[256];
    __shared__ int   sel[KNN];
    int tid = threadIdx.x;
    int q   = blockIdx.x;               // b*Nq + qi
    int b   = q / Nq, qi = q % Nq;
    const float* Q = pq + (long)b * q_bstride + (long)qi * 3;
    float qx = Q[0], qy = Q[1], qz = Q[2];
    const float* C = pc + (long)b * c_bstride;

    for (int j = tid; j < Nc; j += 256) {
        float dx = C[j * 3] - qx, dy = C[j * 3 + 1] - qy, dz = C[j * 3 + 2] - qz;
        d2s[j] = dx * dx + dy * dy + dz * dz;
    }
    __syncthreads();

    for (int slot = 0; slot < KNN; ++slot) {
        float bd = 3.0e38f; int bi = 0x7fffffff;
        for (int j = tid; j < Nc; j += 256) {
            float d = d2s[j];
            if (d < bd || (d == bd && j < bi)) { bd = d; bi = j; }
        }
        rd[tid] = bd; ri[tid] = bi;
        __syncthreads();
        for (int off = 128; off > 0; off >>= 1) {
            if (tid < off) {
                float od = rd[tid + off]; int oi = ri[tid + off];
                if (od < rd[tid] || (od == rd[tid] && oi < ri[tid])) {
                    rd[tid] = od; ri[tid] = oi;
                }
            }
            __syncthreads();
        }
        if (tid == 0) { sel[slot] = ri[0]; d2s[ri[0]] = 3.0e38f; }
        __syncthreads();
    }

    if (tid < KSLOT) {
        long ebase = (long)q * KSLOT + tid;
        if (tid < KNN) {
            int j = sel[tid];
            idx_out[ebase] = j;
            float dx = C[j * 3] - qx, dy = C[j * 3 + 1] - qy, dz = C[j * 3 + 2] - qz;
            float dist = sqrtf(dx * dx + dy * dy + dz * dz + EPSF);
            float inv  = 1.f / dist;
            ev_out[ebase * 3 + 0] = dx * inv;
            ev_out[ebase * 3 + 1] = dy * inv;
            ev_out[ebase * 3 + 2] = dz * inv;
            float d10 = dist * 10.f;
            const float delta = 20.f / 61.f;
            const float coeff = -0.5f / (delta * delta);
            _Float16* E = es_out + ebase * 64;
            E[0] = (_Float16)0.f;                       // d10 >= 0 always
            for (int i = 0; i < 62; ++i) {
                float t = d10 - delta * (float)i;
                E[1 + i] = (_Float16)__expf(coeff * t * t);
            }
            E[63] = (_Float16)((d10 >= 20.f) ? 1.f : 0.f);
        } else {
            idx_out[ebase] = -1;
            for (int c = 0; c < 3; ++c) ev_out[ebase * 3 + c] = 0.f;
            _Float16* E = es_out + ebase * 64;
            for (int i = 0; i < 64; ++i) E[i] = (_Float16)0.f;
        }
    }
}

// ------------------------------------------------------------------
// GVP conv: block = 128 threads = 4 waves = 4 queries x 32 edge slots.
// Vector channels (3% FLOPs) in VALU; scalar channels (97% FLOPs) via
// v_wmma_f32_16x16x32_f16 on 16-edge tiles from an LDS f16 staging tile.
// Both A-tiles are processed per K-chunk so each B-fragment load feeds
// two WMMAs (halves B traffic, enables partial loadcnt waits).
// ------------------------------------------------------------------
struct ConvW {
    const float*    Wh[3];
    const float*    Wmu[3];
    const float*    bs[3];
    const _Float16* Wsp[3];
};

template<int L>
__device__ __forceinline__
void conv_layer(const ConvW& cw, bool active, int w, int lane,
                _Float16 (*stag)[32][224], float (*vred)[32][48],
                float (&mv)[17][3],
                long node, long ctxnode,
                const float* hs_in, const float* hv_in,
                float* hs_out, float* hv_out, int shortcut_add)
{
    constexpr int VI = (L == 0) ? 17 : 16;
    constexpr int H  = (L == 0) ? 17 : 16;
    constexpr int SI = (L == 0) ? 192 : 128;
    constexpr int KC = (L == 0) ? 7 : 5;

    float Vm[16][3];
    if (active) {
        // ---- vector phase: Vh = Wh @ mv, vn -> staging, Vm = Wmu @ Vh ----
        const float* Wh = cw.Wh[L];
        float Vh[H][3];
        #pragma unroll
        for (int hh = 0; hh < H; ++hh) {
            float ax = 0.f, ay = 0.f, az = 0.f;
            #pragma unroll
            for (int v = 0; v < VI; ++v) {
                float wv = Wh[hh * VI + v];         // uniform -> s_load broadcast
                ax += wv * mv[v][0]; ay += wv * mv[v][1]; az += wv * mv[v][2];
            }
            Vh[hh][0] = ax; Vh[hh][1] = ay; Vh[hh][2] = az;
            float vn = sqrtf(ax * ax + ay * ay + az * az + EPSF);
            stag[w][lane][SI + hh] = (_Float16)vn;
        }
        const float* Wmu = cw.Wmu[L];
        #pragma unroll
        for (int o = 0; o < 16; ++o) {
            float ax = 0.f, ay = 0.f, az = 0.f;
            #pragma unroll
            for (int hh = 0; hh < H; ++hh) {
                float wv = Wmu[o * H + hh];
                ax += wv * Vh[hh][0]; ay += wv * Vh[hh][1]; az += wv * Vh[hh][2];
            }
            Vm[o][0] = ax; Vm[o][1] = ay; Vm[o][2] = az;
        }
        if (L < 2) {                                 // vector gate + feed next layer
            #pragma unroll
            for (int o = 0; o < 16; ++o) {
                float nn = sqrtf(Vm[o][0] * Vm[o][0] + Vm[o][1] * Vm[o][1] +
                                 Vm[o][2] * Vm[o][2] + EPSF);
                float sg = 1.f / (1.f + __expf(-nn));
                mv[o][0] = Vm[o][0] * sg;
                mv[o][1] = Vm[o][1] * sg;
                mv[o][2] = Vm[o][2] * sg;
            }
        }
    }
    __syncthreads();

    // ---- scalar phase: fused 2 A-tiles x 8 N-tiles, B shared per K-chunk ----
    float colsum[8];
    #pragma unroll
    for (int nt = 0; nt < 8; ++nt) colsum[nt] = 0.f;

    if (active) {
        const _Float16* Wsp = cw.Wsp[L];
        const float*    bsp = cw.bs[L];
        int m  = lane & 15;
        int hi = lane >> 4;

        v8f acc0[8], acc1[8];
        #pragma unroll
        for (int nt = 0; nt < 8; ++nt) {
            #pragma unroll
            for (int g = 0; g < 8; ++g) { acc0[nt][g] = 0.f; acc1[nt][g] = 0.f; }
        }
        for (int kc = 0; kc < KC; ++kc) {
            const _Float16* ap0 = &stag[w][m][kc * 32 + hi * 8];
            const _Float16* ap1 = &stag[w][16 + m][kc * 32 + hi * 8];
            v8h a0lo = *(const v8h*)ap0;           // K {0..7} or {8..15}
            v8h a0hi = *(const v8h*)(ap0 + 16);    // K {16..23} or {24..31}
            v8h a1lo = *(const v8h*)ap1;
            v8h a1hi = *(const v8h*)(ap1 + 16);
            v16h A0 = __builtin_shufflevector(a0lo, a0hi,
                      0,1,2,3,4,5,6,7,8,9,10,11,12,13,14,15);
            v16h A1 = __builtin_shufflevector(a1lo, a1hi,
                      0,1,2,3,4,5,6,7,8,9,10,11,12,13,14,15);
            #pragma unroll
            for (int nt = 0; nt < 8; ++nt) {
                v16h Bf = *(const v16h*)(Wsp + (((kc << 3) + nt) << 9) + (lane << 4));
                acc0[nt] = __builtin_amdgcn_wmma_f32_16x16x32_f16(
                    false, A0, false, Bf, (short)0, acc0[nt], false, false);
                acc1[nt] = __builtin_amdgcn_wmma_f32_16x16x32_f16(
                    false, A1, false, Bf, (short)0, acc1[nt], false, false);
            }
        }
        #pragma unroll
        for (int nt = 0; nt < 8; ++nt) {
            float bsv = bsp[nt * 16 + m];
            if (L < 2) {
                #pragma unroll
                for (int g = 0; g < 8; ++g) {
                    float x0 = fmaxf(acc0[nt][g] + bsv, 0.f);         // ReLU
                    float x1 = fmaxf(acc1[nt][g] + bsv, 0.f);
                    stag[w][hi * 8 + g][nt * 16 + m]      = (_Float16)x0;
                    stag[w][16 + hi * 8 + g][nt * 16 + m] = (_Float16)x1;
                }
            } else {
                #pragma unroll
                for (int g = 0; g < 8; ++g) {
                    int r0 = hi * 8 + g;                              // edge slot
                    int r1 = 16 + hi * 8 + g;
                    colsum[nt] += (r0 < KNN) ? (acc0[nt][g] + bsv) : 0.f;
                    colsum[nt] += (r1 < KNN) ? (acc1[nt][g] + bsv) : 0.f;
                }
            }
        }
        if (L == 2) {                                 // mean over k + shortcut
            #pragma unroll
            for (int nt = 0; nt < 8; ++nt) {
                float cs = colsum[nt] + __shfl_xor(colsum[nt], 16, 32);
                float as = cs * (1.f / (float)KNN);
                if (shortcut_add) as += hs_in[ctxnode * HID_S + nt * 16 + m];
                if (hi == 0) hs_out[node * HID_S + nt * 16 + m] = as;
            }
        }
    }

    if (L == 2) {                                     // vector mean over k
        __syncthreads();
        if (active) {
            #pragma unroll
            for (int i = 0; i < 48; ++i)
                vred[w][lane][i] = (lane < KNN) ? Vm[i / 3][i % 3] : 0.f;
        }
        __syncthreads();
        if (active) {
            for (int ii = lane; ii < 48; ii += 32) {
                float s = 0.f;
                for (int Lx = 0; Lx < 32; ++Lx) s += vred[w][Lx][ii];
                float av = s * (1.f / (float)KNN);
                if (shortcut_add) av += hv_in[ctxnode * 48 + ii];
                hv_out[node * 48 + ii] = av;
            }
        }
    }
}

__global__ void __launch_bounds__(128)
gvp_conv_kernel(ConvW cw, const float* hs_in, const float* hv_in,
                const int* idx, const _Float16* es, const float* ev,
                float* hs_out, float* hv_out,
                int Nq, int Nctx, int first, int shortcut)
{
    __shared__ __align__(16) _Float16 stag[4][32][224];   // 56 KB staging
    __shared__ float vred[4][32][48];                     // 24 KB vec reduce
    int tid  = threadIdx.x;
    int w    = tid >> 5;
    int lane = tid & 31;
    long q = (long)blockIdx.x * 4 + w;
    bool active = q < (long)BATCH * Nq;

    long node = 0, ctxnode = 0;
    float mv[17][3];
    #pragma unroll
    for (int o = 0; o < 17; ++o) { mv[o][0] = mv[o][1] = mv[o][2] = 0.f; }

    if (active) {
        long b = q / Nq, n = q % Nq;
        node = q;
        ctxnode = b * Nctx + n;
        long ebase = q * KSLOT + lane;
        int j = idx[ebase];
        bool valid = (j >= 0);

        // stage scalar cols [0,128): gathered sender features (f16)
        if (!first && valid) {
            const float* hp = hs_in + (b * (long)Nctx + j) * HID_S;
            #pragma unroll
            for (int k = 0; k < HID_S; k += 4) {
                float4 f = *(const float4*)(hp + k);
                v4h hh;
                hh[0] = (_Float16)f.x; hh[1] = (_Float16)f.y;
                hh[2] = (_Float16)f.z; hh[3] = (_Float16)f.w;
                *(v4h*)&stag[w][lane][k] = hh;
            }
        } else {
            v4h z; z[0] = z[1] = z[2] = z[3] = (_Float16)0.f;
            #pragma unroll
            for (int k = 0; k < HID_S; k += 4) *(v4h*)&stag[w][lane][k] = z;
        }
        // stage edge RBF cols [128,192)
        const _Float16* ep = es + ebase * 64;
        #pragma unroll
        for (int k = 0; k < 64; k += 4)
            *(v4h*)&stag[w][lane][128 + k] = *(const v4h*)(ep + k);

        // vector regs: gathered hv (16) + edge dir (1)
        if (!first && valid) {
            const float* vp = hv_in + (b * (long)Nctx + j) * 48;
            #pragma unroll
            for (int o = 0; o < 16; ++o) {
                mv[o][0] = vp[o * 3]; mv[o][1] = vp[o * 3 + 1]; mv[o][2] = vp[o * 3 + 2];
            }
        }
        if (valid) {
            mv[16][0] = ev[ebase * 3];
            mv[16][1] = ev[ebase * 3 + 1];
            mv[16][2] = ev[ebase * 3 + 2];
        }
    }

    int sc = (shortcut && !first) ? 1 : 0;
    conv_layer<0>(cw, active, w, lane, stag, vred, mv, node, ctxnode,
                  hs_in, hv_in, hs_out, hv_out, sc);
    conv_layer<1>(cw, active, w, lane, stag, vred, mv, node, ctxnode,
                  hs_in, hv_in, hs_out, hv_out, sc);
    conv_layer<2>(cw, active, w, lane, stag, vred, mv, node, ctxnode,
                  hs_in, hv_in, hs_out, hv_out, sc);
}

// ------------------------------------------------------------------
// Out GVP (vector branch only) + Gram-Schmidt frames per node.
// ------------------------------------------------------------------
__global__ void frames_kernel(const float* Wh, const float* Wmu, const float* hv,
                              float* Rout, int M)
{
    int t = blockIdx.x * blockDim.x + threadIdx.x;
    if (t >= BATCH * M) return;
    const float* vp = hv + (long)t * 48;
    float Vh[16][3];
    #pragma unroll
    for (int hh = 0; hh < 16; ++hh) {
        float ax = 0.f, ay = 0.f, az = 0.f;
        #pragma unroll
        for (int v = 0; v < 16; ++v) {
            float wv = Wh[hh * 16 + v];
            ax += wv * vp[v * 3]; ay += wv * vp[v * 3 + 1]; az += wv * vp[v * 3 + 2];
        }
        Vh[hh][0] = ax; Vh[hh][1] = ay; Vh[hh][2] = az;
    }
    float Vm[8][3];
    #pragma unroll
    for (int o = 0; o < 8; ++o) {
        float ax = 0.f, ay = 0.f, az = 0.f;
        #pragma unroll
        for (int hh = 0; hh < 16; ++hh) {
            float wv = Wmu[o * 16 + hh];
            ax += wv * Vh[hh][0]; ay += wv * Vh[hh][1]; az += wv * Vh[hh][2];
        }
        Vm[o][0] = ax; Vm[o][1] = ay; Vm[o][2] = az;
    }
    #pragma unroll
    for (int f = 0; f < NF; ++f) {
        float v1x = Vm[2 * f][0], v1y = Vm[2 * f][1], v1z = Vm[2 * f][2];
        float v2x = Vm[2 * f + 1][0], v2y = Vm[2 * f + 1][1], v2z = Vm[2 * f + 1][2];
        float n1 = sqrtf(v1x * v1x + v1y * v1y + v1z * v1z + EPSF);
        float e1x = v1x / n1, e1y = v1y / n1, e1z = v1z / n1;
        float d = e1x * v2x + e1y * v2y + e1z * v2z;
        float u2x = v2x - d * e1x, u2y = v2y - d * e1y, u2z = v2z - d * e1z;
        float n2 = sqrtf(u2x * u2x + u2y * u2y + u2z * u2z + EPSF);
        float e2x = u2x / n2, e2y = u2y / n2, e2z = u2z / n2;
        float e3x = e1y * e2z - e1z * e2y;
        float e3y = e1z * e2x - e1x * e2z;
        float e3z = e1x * e2y - e1y * e2x;
        float* o = Rout + ((long)t * NF + f) * 9;   // R[i][col], cols = e1,e2,e3
        o[0] = e1x; o[1] = e2x; o[2] = e3x;
        o[3] = e1y; o[4] = e2y; o[5] = e3y;
        o[6] = e1z; o[7] = e2z; o[8] = e3z;
    }
}

// ------------------------------------------------------------------
// Nearest-neighbor unpool of (M,36) frames into d_out (B,N,36,3) slab.
// ------------------------------------------------------------------
__global__ void unpool_kernel(const float* pthis, long t_bstride, int M,
                              const float* p, const float* Rbuf, float* outR, int li)
{
    int t = blockIdx.x * 256 + threadIdx.x;   // b*NPTS + n
    if (t >= BATCH * NPTS) return;
    int b = t / NPTS;
    float qx = p[(long)t * 3], qy = p[(long)t * 3 + 1], qz = p[(long)t * 3 + 2];
    const float* C = pthis + (long)b * t_bstride;
    float bd = 3.0e38f; int bi = 0;
    for (int jm = 0; jm < M; ++jm) {
        float dx = C[jm * 3] - qx, dy = C[jm * 3 + 1] - qy, dz = C[jm * 3 + 2] - qz;
        float d2 = dx * dx + dy * dy + dz * dz;
        if (d2 < bd) { bd = d2; bi = jm; }
    }
    const float* src = Rbuf + ((long)b * M + bi) * 36;
    float* dst = outR + (long)t * 108 + li * 36;
    for (int r = 0; r < 36; ++r) dst[r] = src[r];
}

__global__ void copy_kernel(const float* src, float* dst, int n)
{
    int t = blockIdx.x * 256 + threadIdx.x;
    if (t < n) dst[t] = src[t];
}

// ------------------------------------------------------------------
// Host orchestration.
// Input flattening (jax pytree, sorted dict keys):
//   0: p; encode stack s layer l at 1+s*12+l*4 (Wh,Wmu,Ws,bs);
//   out li at 25+li*4; scale s layer l at 37+s*12+l*4.
// ------------------------------------------------------------------
extern "C" void kernel_launch(void* const* d_in, const int* in_sizes, int n_in,
                              void* d_out, int out_size, void* d_ws, size_t ws_size,
                              hipStream_t stream)
{
    (void)in_sizes; (void)n_in; (void)out_size; (void)ws_size;
    const float* p = (const float*)d_in[0];
    auto IN = [&](int i) { return (const float*)d_in[i]; };

    char* W = (char*)d_ws;
    size_t off = 0;
    _Float16* wpack = (_Float16*)(W + off); off += (size_t)5 * 69632 * sizeof(_Float16);
    float* hs0 = (float*)(W + off); off += (size_t)BATCH * NPTS * HID_S * 4;
    float* hs1 = (float*)(W + off); off += (size_t)BATCH * NPTS * HID_S * 4;
    float* hv0 = (float*)(W + off); off += (size_t)BATCH * NPTS * 48 * 4;
    float* hv1 = (float*)(W + off); off += (size_t)BATCH * NPTS * 48 * 4;
    int*      idxb = (int*)(W + off);      off += (size_t)BATCH * NPTS * KSLOT * 4;
    _Float16* esb  = (_Float16*)(W + off); off += (size_t)BATCH * NPTS * KSLOT * 64 * 2;
    float*    evb  = (float*)(W + off);    off += (size_t)BATCH * NPTS * KSLOT * 3 * 4;
    float*    pm   = (float*)(W + off);    off += 256;
    float*    Rbuf = (float*)(W + off);    off += (size_t)BATCH * 1024 * 36 * 4;

    // pack all 15 Ws matrices to f16 B-fragment layout
    PackArgs pa;
    for (int s = 0; s < 2; ++s)
        for (int l = 0; l < 3; ++l) pa.Ws[s * 3 + l] = IN(1 + s * 12 + l * 4 + 2);
    for (int s = 0; s < 3; ++s)
        for (int l = 0; l < 3; ++l) pa.Ws[(2 + s) * 3 + l] = IN(37 + s * 12 + l * 4 + 2);
    pack_ws_kernel<<<15, 256, 0, stream>>>(pa, wpack);
    mean_kernel<<<1, 128, 0, stream>>>(p, pm);

    auto mkConv = [&](int baseIdx, int stackPackIdx) {
        ConvW c;
        for (int l = 0; l < 3; ++l) {
            c.Wh[l]  = IN(baseIdx + l * 4 + 0);
            c.Wmu[l] = IN(baseIdx + l * 4 + 1);
            c.bs[l]  = IN(baseIdx + l * 4 + 3);
            c.Wsp[l] = wpack + (long)stackPackIdx * 69632 +
                       (l == 0 ? 0 : (l == 1 ? 28672 : 49152));
        }
        return c;
    };
    ConvW enc0 = mkConv(1, 0),  enc1 = mkConv(13, 1);
    ConvW sc0  = mkConv(37, 2), sc1  = mkConv(49, 3), sc2 = mkConv(61, 4);

    const long pstride = (long)NPTS * 3;

    // ---- encode: 2 residual convs on the full 4096-point graph ----
    knn_kernel<<<BATCH * NPTS, 256, 0, stream>>>(p, pstride, NPTS, p, pstride, NPTS,
                                                 idxb, esb, evb);
    int nblk = (BATCH * NPTS + 3) / 4;
    gvp_conv_kernel<<<nblk, 128, 0, stream>>>(enc0, hs0, hv0, idxb, esb, evb,
                                              hs1, hv1, NPTS, NPTS, 1, 1);
    gvp_conv_kernel<<<nblk, 128, 0, stream>>>(enc1, hs1, hv1, idxb, esb, evb,
                                              hs0, hv0, NPTS, NPTS, 0, 1);
    float* feat_out = (float*)d_out + (size_t)BATCH * NPTS * 108;
    copy_kernel<<<(BATCH * NPTS * HID_S + 255) / 256, 256, 0, stream>>>(
        hs0, feat_out, BATCH * NPTS * HID_S);

    // ---- scale 0: 4096 -> 1024 ----
    knn_kernel<<<BATCH * 1024, 256, 0, stream>>>(p, pstride, NPTS, p, pstride, 1024,
                                                 idxb, esb, evb);
    gvp_conv_kernel<<<(BATCH * 1024 + 3) / 4, 128, 0, stream>>>(
        sc0, hs0, hv0, idxb, esb, evb, hs1, hv1, 1024, NPTS, 0, 0);
    frames_kernel<<<(BATCH * 1024 + 255) / 256, 256, 0, stream>>>(
        IN(25), IN(26), hv1, Rbuf, 1024);
    unpool_kernel<<<(BATCH * NPTS + 255) / 256, 256, 0, stream>>>(
        p, pstride, 1024, p, Rbuf, (float*)d_out, 0);

    // ---- scale 1: 1024 -> 256 ----
    knn_kernel<<<BATCH * 256, 256, 0, stream>>>(p, pstride, 1024, p, pstride, 256,
                                                idxb, esb, evb);
    gvp_conv_kernel<<<(BATCH * 256 + 3) / 4, 128, 0, stream>>>(
        sc1, hs1, hv1, idxb, esb, evb, hs0, hv0, 256, 1024, 0, 0);
    frames_kernel<<<(BATCH * 256 + 255) / 256, 256, 0, stream>>>(
        IN(29), IN(30), hv0, Rbuf, 256);
    unpool_kernel<<<(BATCH * NPTS + 255) / 256, 256, 0, stream>>>(
        p, pstride, 256, p, Rbuf, (float*)d_out, 1);

    // ---- scale 2: 256 -> 1 (centroid query) ----
    knn_kernel<<<BATCH, 256, 0, stream>>>(p, pstride, 256, pm, 3, 1, idxb, esb, evb);
    gvp_conv_kernel<<<1, 128, 0, stream>>>(
        sc2, hs0, hv0, idxb, esb, evb, hs1, hv1, 1, 256, 0, 0);
    frames_kernel<<<1, 256, 0, stream>>>(IN(33), IN(34), hv1, Rbuf, 1);
    unpool_kernel<<<(BATCH * NPTS + 255) / 256, 256, 0, stream>>>(
        pm, 3, 1, p, Rbuf, (float*)d_out, 2);
}
